// DM_fixed_k_44504451121741
// MI455X (gfx1250) — matrix operand order repaired and checked by reading, not compiled, not run on hardware
//
#include <hip/hip_runtime.h>
#include <hip/hip_bf16.h>

#define B_    8
#define S_    4096
#define D_    256
#define H_    1024
#define KSEL  2048
#define KK    2047
#define TOKTILES 128   // KSEL/16

typedef __bf16 bf16_t;
typedef bf16_t v16bf __attribute__((ext_vector_type(16)));
typedef float  v8f   __attribute__((ext_vector_type(8)));

union FragU { uint4 u[2]; v16bf v; };

__device__ __forceinline__ unsigned short f2bf(float f) {
  union { float f; unsigned int u; } c; c.f = f;
  unsigned int u = c.u;
  u += 0x7FFFu + ((u >> 16) & 1u);   // round-to-nearest-even
  return (unsigned short)(u >> 16);
}

// branch-free tanh-GELU: native v_tanh_f32 if available, else exp-based
__device__ __forceinline__ float gelu_tanh(float v) {
  float z = 0.7978845608028654f * (v + 0.044715f * v * v * v);
#if __has_builtin(__builtin_amdgcn_tanhf)
  float t = __builtin_amdgcn_tanhf(z);
#else
  float e = __expf(2.0f * z);
  float t = 1.0f - 2.0f / (e + 1.0f);
#endif
  return 0.5f * v * (1.0f + t);
}

// async copy: global -> LDS, tracked by ASYNCcnt (CDNA5 async path)
__device__ __forceinline__ void async_load_b128(unsigned lds_off, const void* gptr) {
  asm volatile("global_load_async_to_lds_b128 %0, %1, off"
               :: "v"(lds_off), "v"(gptr) : "memory");
}
__device__ __forceinline__ void wait_asynccnt0() {
  asm volatile("s_wait_asynccnt 0x0" ::: "memory");
}

// ---------------- router: w[b,s] = x[b,s,:] . wr + br ----------------
__global__ __launch_bounds__(256)
void router_kernel(const float* __restrict__ x, const float* __restrict__ wr,
                   const float* __restrict__ br, float* __restrict__ weights) {
  int wave = threadIdx.x >> 5;
  int lane = threadIdx.x & 31;
  int tok  = blockIdx.x * 8 + wave;                 // 0..B*S-1
  const float4* xp = (const float4*)(x + (size_t)tok * D_) + lane * 2;
  const float4* wp = (const float4*)(wr) + lane * 2;
  float4 a0 = xp[0], a1 = xp[1];
  float4 b0 = wp[0], b1 = wp[1];
  float s = a0.x*b0.x + a0.y*b0.y + a0.z*b0.z + a0.w*b0.w
          + a1.x*b1.x + a1.y*b1.y + a1.z*b1.z + a1.w*b1.w;
  #pragma unroll
  for (int off = 16; off > 0; off >>= 1) s += __shfl_xor(s, off, 32);
  if (lane == 0) weights[tok] = s + br[0];
}

// ---------------- init scratch state (must run every launch) ----------------
__global__ void init_kernel(int* counts, int* sel_idx, float* sel_w) {
  int t = blockIdx.x * blockDim.x + threadIdx.x;
  if (t < B_) counts[t] = 0;
  if (t < B_ * KSEL) { sel_idx[t] = -1; sel_w[t] = 0.f; }
}

// ---- pre-swizzle w1/w2 into bf16 WMMA B-fragment layout ----
// B (32x16 bf16) fragment: lane L -> column n = nt*16 + (L&15),
// 16 consecutive K values starting at kt*32 + (L>>4)*16, stored contiguously
// (32 bytes per lane) so the GEMM loop loads with plain b128s.
__global__ __launch_bounds__(256)
void prep_w_kernel(const float* __restrict__ w1, const float* __restrict__ w2,
                   unsigned short* __restrict__ w1f, unsigned short* __restrict__ w2f) {
  int g = blockIdx.x * blockDim.x + threadIdx.x;   // 0..32767
  int which = g >> 14;
  int r     = g & 16383;
  int lane  = r & 31;
  int frag  = r >> 5;
  int nsub  = lane & 15, half = lane >> 4;
  if (which == 0) {             // w1: K=256 (8 ktiles), N=1024 (64 ntiles)
    int nt = frag >> 3, kt = frag & 7;
    int n  = nt * 16 + nsub;
    int k0 = kt * 32 + half * 16;
    unsigned short* dst = w1f + (size_t)frag * 512 + lane * 16;
    #pragma unroll
    for (int j = 0; j < 16; ++j) dst[j] = f2bf(w1[(size_t)(k0 + j) * H_ + n]);
  } else {                      // w2: K=1024 (32 ktiles), N=256 (16 ntiles)
    int nt = frag >> 5, kt = frag & 31;
    int n  = nt * 16 + nsub;
    int k0 = kt * 32 + half * 16;
    unsigned short* dst = w2f + (size_t)frag * 512 + lane * 16;
    #pragma unroll
    for (int j = 0; j < 16; ++j) dst[j] = f2bf(w2[(size_t)(k0 + j) * D_ + n]);
  }
}

// ---------------- exact top-KK selection via rank counting ----------------
__global__ __launch_bounds__(256)
void select_kernel(const float* __restrict__ weights, int* counts,
                   int* sel_idx, float* sel_w) {
  __shared__ float w[S_];
  int b = blockIdx.y;
  const float* wb = weights + (size_t)b * S_;
  for (int i = threadIdx.x; i < S_; i += blockDim.x) w[i] = wb[i];
  __syncthreads();
  int t = blockIdx.x * blockDim.x + threadIdx.x;    // token in batch
  float wt = w[t];
  int rank = 0;
  const float4* w4 = (const float4*)w;
  for (int j4 = 0; j4 < S_ / 4; ++j4) {
    float4 v = w4[j4];
    int j = j4 * 4;
    rank += (v.x > wt) || (v.x == wt && (j + 0) < t);
    rank += (v.y > wt) || (v.y == wt && (j + 1) < t);
    rank += (v.z > wt) || (v.z == wt && (j + 2) < t);
    rank += (v.w > wt) || (v.w == wt && (j + 3) < t);
  }
  if (rank < KK) {                                  // token is in top-2047
    int slot = atomicAdd(&counts[b], 1);
    sel_idx[b * KSEL + slot] = t;
    sel_w[b * KSEL + slot]  = wt;
  }
}

// ---------------- fused gather -> GEMM1+GELU -> GEMM2 -> scaled scatter ----
#define XS 264    // xa row stride (ushorts): 528B, 16B aligned, odd dword phase
#define HS 1032   // h  row stride (ushorts): 2064B
#define YS 260    // y  row stride (floats)

__global__ __launch_bounds__(256)
void ffn_kernel(const float* __restrict__ x, const float* __restrict__ b1,
                const float* __restrict__ b2, const unsigned short* __restrict__ w1f,
                const unsigned short* __restrict__ w2f, const int* __restrict__ sel_idx,
                const float* __restrict__ sel_w, float* __restrict__ out) {
  __shared__ __align__(16) unsigned short xa[16 * XS];
  __shared__ __align__(16) unsigned short hl[16 * HS];
  __shared__ __align__(16) float          yxbuf[16 * YS];  // xf staging, then yl
  __shared__ int   ridx[16];
  __shared__ float rw[16];
  float* xf = yxbuf;   // phase 1: raw f32 rows (16 x 256)
  float* yl = yxbuf;   // phase 3: y tile (16 x YS)

  int b    = blockIdx.x >> 7;        // TOKTILES = 128 tiles per batch
  int tile = blockIdx.x & 127;
  int tid  = threadIdx.x;
  int lane = tid & 31;
  int wave = tid >> 5;

  if (tid < 16) {
    ridx[tid] = sel_idx[b * KSEL + tile * 16 + tid];
    rw[tid]   = sel_w [b * KSEL + tile * 16 + tid];
  }
  __syncthreads();

  // stage raw f32 rows into LDS with the async copy engine (ASYNCcnt path)
  {
    int r  = tid >> 4;
    int c0 = (tid & 15) * 16;                       // float index within row
    int idx = ridx[r];
    if (idx >= 0) {
      const float* src = x + ((size_t)b * S_ + idx) * D_ + c0;
      unsigned dst = (unsigned)(size_t)&xf[r * 256 + c0];  // LDS byte offset
      #pragma unroll
      for (int q = 0; q < 4; ++q)
        async_load_b128(dst + q * 16, src + q * 4);
    }
    wait_asynccnt0();
  }
  __syncthreads();

  // convert f32 -> bf16 into WMMA-friendly padded layout
  {
    int r  = tid >> 4;
    int c0 = (tid & 15) * 16;
    int idx = ridx[r];
    unsigned short* dst = &xa[r * XS + c0];
    if (idx >= 0) {
      const float4* src = (const float4*)&xf[r * 256 + c0];
      #pragma unroll
      for (int q = 0; q < 4; ++q) {
        float4 v = src[q];
        dst[q * 4 + 0] = f2bf(v.x); dst[q * 4 + 1] = f2bf(v.y);
        dst[q * 4 + 2] = f2bf(v.z); dst[q * 4 + 3] = f2bf(v.w);
      }
    } else {
      #pragma unroll
      for (int q = 0; q < 16; ++q) dst[q] = 0;
    }
  }
  __syncthreads();

  int m = lane & 15, half = lane >> 4;

  // ---- GEMM1: h(16x1024) = gelu(xa(16x256) @ w1 + b1); 8 N-tiles per wave
  for (int tn = 0; tn < 8; ++tn) {
    int nt = wave * 8 + tn;
    v8f acc = {};
    #pragma unroll
    for (int kt = 0; kt < 8; ++kt) {
      FragU A, Bf;
      // A fragment: K = kt*32 + half*16 .. 16 contiguous (matches B packing)
      A.u[0] = *(const uint4*)&xa[m * XS + kt * 32 + half * 16];
      A.u[1] = *(const uint4*)&xa[m * XS + kt * 32 + half * 16 + 8];
      const uint4* bp = (const uint4*)(w1f + (size_t)(nt * 8 + kt) * 512 + lane * 16);
      Bf.u[0] = bp[0]; Bf.u[1] = bp[1];
      if (kt < 7) __builtin_prefetch(bp + 32, 0, 3);   // 512 B ahead (half a fragment), near scope
      acc = __builtin_amdgcn_wmma_f32_16x16x32_bf16(false, A.v, false, Bf.v,
                                                    (short)0, acc, false, false);
    }
    int n = nt * 16 + (lane & 15);
    float bias = b1[n];
    #pragma unroll
    for (int r = 0; r < 8; ++r) {           // C layout: row = r + 8*half, col = n
      float g = gelu_tanh(acc[r] + bias);
      hl[(r + 8 * half) * HS + n] = f2bf(g);
    }
  }
  __syncthreads();

  // ---- GEMM2: y(16x256) = h(16x1024) @ w2; 2 N-tiles per wave
  for (int tn = 0; tn < 2; ++tn) {
    int nt = wave * 2 + tn;
    v8f acc = {};
    #pragma unroll 8
    for (int kt = 0; kt < 32; ++kt) {
      FragU A, Bf;
      A.u[0] = *(const uint4*)&hl[m * HS + kt * 32 + half * 16];
      A.u[1] = *(const uint4*)&hl[m * HS + kt * 32 + half * 16 + 8];
      const uint4* bp = (const uint4*)(w2f + (size_t)(nt * 32 + kt) * 512 + lane * 16);
      Bf.u[0] = bp[0]; Bf.u[1] = bp[1];
      acc = __builtin_amdgcn_wmma_f32_16x16x32_bf16(false, A.v, false, Bf.v,
                                                    (short)0, acc, false, false);
    }
    int n = nt * 16 + (lane & 15);
    #pragma unroll
    for (int r = 0; r < 8; ++r) yl[(r + 8 * half) * YS + n] = acc[r];
  }
  __syncthreads();

  // ---- epilogue: out[b, idx, :] = x + (y + b2) * w_sel   (coalesced rows)
  {
    float bias = b2[tid];
    for (int r = 0; r < 16; ++r) {
      int idx = ridx[r];
      if (idx < 0) continue;
      size_t o = ((size_t)b * S_ + idx) * D_ + tid;
      out[o] = x[o] + (yl[r * YS + tid] + bias) * rw[r];
    }
  }
}

extern "C" void kernel_launch(void* const* d_in, const int* in_sizes, int n_in,
                              void* d_out, int out_size, void* d_ws, size_t ws_size,
                              hipStream_t stream) {
  const float* x  = (const float*)d_in[0];
  // d_in[1] = attention_mask (all zeros; unused by the math)
  const float* wr = (const float*)d_in[2];
  const float* br = (const float*)d_in[3];
  const float* w1 = (const float*)d_in[4];
  const float* b1 = (const float*)d_in[5];
  const float* w2 = (const float*)d_in[6];
  const float* b2 = (const float*)d_in[7];
  float* out = (float*)d_out;

  char* p = (char*)d_ws;
  float* weights       = (float*)p;           p += (size_t)B_ * S_ * sizeof(float);
  int*   counts        = (int*)p;             p += 256;
  int*   sel_idx       = (int*)p;             p += (size_t)B_ * KSEL * sizeof(int);
  float* sel_w         = (float*)p;           p += (size_t)B_ * KSEL * sizeof(float);
  unsigned short* w1f  = (unsigned short*)p;  p += (size_t)D_ * H_ * sizeof(unsigned short);
  unsigned short* w2f  = (unsigned short*)p;  p += (size_t)H_ * D_ * sizeof(unsigned short);

  // unselected tokens pass through: out = x, FFN overwrites selected rows
  hipMemcpyAsync(d_out, d_in[0], (size_t)B_ * S_ * D_ * sizeof(float),
                 hipMemcpyDeviceToDevice, stream);
  init_kernel<<<64, 256, 0, stream>>>(counts, sel_idx, sel_w);
  router_kernel<<<B_ * S_ / 8, 256, 0, stream>>>(x, wr, br, weights);
  prep_w_kernel<<<128, 256, 0, stream>>>(w1, w2, w1f, w2f);
  select_kernel<<<dim3(16, 8), 256, 0, stream>>>(weights, counts, sel_idx, sel_w);
  ffn_kernel<<<B_ * TOKTILES, 256, 0, stream>>>(x, b1, b2, w1f, w2f, sel_idx, sel_w, out);
}
